// OptimizedQwen3VLAttention_6983616824471
// MI455X (gfx1250) — matrix operand order repaired and checked by reading, not compiled, not run on hardware
//
#include <hip/hip_runtime.h>
#include <hip/hip_bf16.h>

// ---------------------------------------------------------------------------
// Types for CDNA5 WMMA (wave32): v_wmma_f32_16x16x32_bf16
// ---------------------------------------------------------------------------
typedef __attribute__((ext_vector_type(16))) __bf16 bf16x16;
typedef __attribute__((ext_vector_type(8)))  __bf16 bf16x8;
typedef __attribute__((ext_vector_type(8)))  float  f32x8;

static __device__ __forceinline__ bf16x16 make16(bf16x8 lo, bf16x8 hi) {
    union { bf16x16 v; bf16x8 h[2]; } u;
    u.h[0] = lo; u.h[1] = hi;
    return u.v;
}

// A-fragment (16x32 bf16, MxK): lanes 0-15 -> M=lane, K={0..7,16..23};
// lanes 16-31 -> M=lane-16, K={8..15,24..31}. Source [m][k], row stride ldm.
static __device__ __forceinline__ bf16x16 frag_a(const __bf16* base, int ldm, int lane) {
    int m = lane & 15, half = lane >> 4;
    const __bf16* p = base + (size_t)m * ldm + half * 8;
    return make16(*(const bf16x8*)p, *(const bf16x8*)(p + 16));
}

// B-fragment (32x16 bf16, KxN): lanes 0-15 -> N=lane, K={0..7,16..23}; lanes 16-31 ->
// N=lane-16, K={8..15,24..31}. Source [n][k] (K-contiguous per column), row stride ldn.
static __device__ __forceinline__ bf16x16 frag_b(const __bf16* base, int ldn, int lane) {
    int n = lane & 15, half = lane >> 4;
    const __bf16* p = base + (size_t)n * ldn + half * 8;
    return make16(*(const bf16x8*)p, *(const bf16x8*)(p + 16));
}

static __device__ __forceinline__ f32x8 wmma_bf16(bf16x16 a, bf16x16 b, f32x8 c) {
    return __builtin_amdgcn_wmma_f32_16x16x32_bf16(false, a, false, b, (short)0, c, false, false);
}

// ---------------------------------------------------------------------------
// CDNA5 async global->LDS copy (ASYNCcnt-tracked DMA, no VGPR staging).
// ---------------------------------------------------------------------------
static __device__ __forceinline__ unsigned lds_addr(const void* p) {
    return (unsigned)(size_t)p;   // low 32 bits of generic LDS address = LDS offset
}
static __device__ __forceinline__ void async_copy_b128(unsigned ldsoff, const void* gsrc) {
    asm volatile("global_load_async_to_lds_b128 %0, %1, off"
                 :: "v"(ldsoff), "v"(gsrc) : "memory");
}
#define WAIT_ASYNC(n) asm volatile("s_wait_asynccnt %0" :: "n"(n) : "memory")

// ---------------------------------------------------------------------------
// Kernel 1: f32 -> bf16 conversion (grid-stride, 8 elems/thread)
// ---------------------------------------------------------------------------
__global__ __launch_bounds__(256) void cvt_f32_bf16(
    const float* __restrict__ x, __bf16* __restrict__ y, long n) {
    long i = ((long)blockIdx.x * blockDim.x + threadIdx.x) * 8;
    long stride = (long)gridDim.x * blockDim.x * 8;
    for (; i < n; i += stride) {
        bf16x8 o;
#pragma unroll
        for (int j = 0; j < 8; ++j) o[j] = (__bf16)x[i + j];
        *(bf16x8*)&y[i] = o;
    }
}

// ---------------------------------------------------------------------------
// Kernel 2: 64x64 LDS-tiled transpose with f32->bf16 convert.
// X is (R x C) f32 with row stride xrs; Y is (C x R) bf16 with row stride yrs.
// Per-z bases: X += (z/zmod)*xos + (z%zmod)*xis ; Y += z*ybs.
// ---------------------------------------------------------------------------
__global__ __launch_bounds__(256) void transpose_cvt(
    const float* __restrict__ X, __bf16* __restrict__ Y,
    long xrs, long yrs, long xos, long xis, int zmod, long ybs) {
    __shared__ float t[64][65];
    int z = blockIdx.z;
    const float* xb = X + (size_t)(z / zmod) * xos + (size_t)(z % zmod) * xis;
    __bf16* yb = Y + (size_t)z * ybs;
    int r0 = blockIdx.y * 64, c0 = blockIdx.x * 64;
    int tc = threadIdx.x & 63, tr = threadIdx.x >> 6;  // tr in 0..3
#pragma unroll
    for (int i = 0; i < 64; i += 4)
        t[tr + i][tc] = xb[(size_t)(r0 + tr + i) * xrs + c0 + tc];
    __syncthreads();
#pragma unroll
    for (int i = 0; i < 64; i += 4)
        yb[(size_t)(c0 + tr + i) * yrs + r0 + tc] = (__bf16)t[tc][tr + i];
}

// ---------------------------------------------------------------------------
// Kernel 3: bf16 WMMA GEMM.  C[MxN] = A[MxK] * Bt^T, A row-major, Bt is the
// pre-transposed B (N x K row-major), C f32 row-major.
// Block tile 128x128x32, 8 waves, double-buffered async global->LDS copies,
// last K-tile peeled so the steady-state loop is branch-free.
// ---------------------------------------------------------------------------
#define BM 128
#define BN 128
#define BK 32

__global__ __launch_bounds__(256) void gemm_bf16(
    const __bf16* __restrict__ A, const __bf16* __restrict__ Bt,
    float* __restrict__ C, int M, int N, int K) {
    __shared__ __bf16 As[2][BM][BK];   // [m][k]
    __shared__ __bf16 Bs[2][BN][BK];   // [n][k]

    const int tid = threadIdx.x;
    const int lane = tid & 31, wave = tid >> 5;
    const int waveM = wave >> 2, waveN = wave & 3;   // 2x4 wave grid
    const int bm = blockIdx.y * BM, bn = blockIdx.x * BN;

    // fixed per-thread copy slots: rows (tid>>2) and (tid>>2)+64, k-slot (tid&3)*8
    const int cm = tid >> 2, ck = (tid & 3) * 8;
    const __bf16* ap0 = A  + (size_t)(bm + cm) * K + ck;
    const __bf16* ap1 = ap0 + (size_t)64 * K;
    const __bf16* bp0 = Bt + (size_t)(bn + cm) * K + ck;
    const __bf16* bp1 = bp0 + (size_t)64 * K;
    const unsigned la0 = lds_addr(&As[0][cm][ck]);
    const unsigned la1 = lds_addr(&As[0][cm + 64][ck]);
    const unsigned lb0 = lds_addr(&Bs[0][cm][ck]);
    const unsigned lb1 = lds_addr(&Bs[0][cm + 64][ck]);
    const unsigned bufstep = (unsigned)(BM * BK * sizeof(__bf16));  // 8192

    auto issue = [&](int k0, int buf) {
        unsigned o = buf ? bufstep : 0u;
        async_copy_b128(la0 + o, ap0 + k0);
        async_copy_b128(la1 + o, ap1 + k0);
        async_copy_b128(lb0 + o, bp0 + k0);
        async_copy_b128(lb1 + o, bp1 + k0);
    };

    f32x8 acc[4][2] = {};
    auto compute = [&](int cur) {
        bf16x16 bf[2];
#pragma unroll
        for (int nt = 0; nt < 2; ++nt)
            bf[nt] = frag_b(&Bs[cur][waveN * 32 + nt * 16][0], BK, lane);
#pragma unroll
        for (int mt = 0; mt < 4; ++mt) {
            bf16x16 af = frag_a(&As[cur][waveM * 64 + mt * 16][0], BK, lane);
#pragma unroll
            for (int nt = 0; nt < 2; ++nt)
                acc[mt][nt] = wmma_bf16(af, bf[nt], acc[mt][nt]);
        }
    };

    issue(0, 0);
    int cur = 0;
    int k0 = 0;
    for (; k0 + BK < K; k0 += BK) {          // steady state: branch-free body
        issue(k0 + BK, cur ^ 1);             // next tile into the other buffer
        WAIT_ASYNC(4);                       // oldest 4 (current tile) complete
        __syncthreads();
        compute(cur);
        __syncthreads();
        cur ^= 1;
    }
    WAIT_ASYNC(0);                           // peeled final tile
    __syncthreads();
    compute(cur);

    const int lm = (lane >> 4) * 8;
    const int ln = lane & 15;
#pragma unroll
    for (int mt = 0; mt < 4; ++mt)
#pragma unroll
        for (int nt = 0; nt < 2; ++nt) {
            int row0 = bm + waveM * 64 + mt * 16 + lm;
            int col  = bn + waveN * 32 + nt * 16 + ln;
#pragma unroll
            for (int r = 0; r < 8; ++r)
                C[(size_t)(row0 + r) * N + col] = acc[mt][nt][r];
        }
}

// ---------------------------------------------------------------------------
// Kernel 4: per-head RMSNorm + RoPE, f32 in -> bf16 out (wave per head-vector)
// ---------------------------------------------------------------------------
__global__ __launch_bounds__(256) void norm_rope(
    const float* __restrict__ X, const float* __restrict__ w,
    const float* __restrict__ cosb, const float* __restrict__ sinb,
    __bf16* __restrict__ Y, int S_, int nheads) {
    int wid  = blockIdx.x * 8 + (threadIdx.x >> 5);
    int lane = threadIdx.x & 31;
    int tok  = wid / nheads;
    int s    = tok % S_;
    const float* x = X + (size_t)wid * 128;

    float v[4]; float ss = 0.f;
#pragma unroll
    for (int j = 0; j < 4; ++j) { v[j] = x[lane + 32 * j]; ss += v[j] * v[j]; }
#pragma unroll
    for (int off = 16; off > 0; off >>= 1) ss += __shfl_xor(ss, off, 32);
    float inv = rsqrtf(ss * (1.f / 128.f) + 1e-6f);

    float xn[4];
#pragma unroll
    for (int j = 0; j < 4; ++j) {
        int d = lane + 32 * j;
        xn[j] = v[j] * inv * w[d];
    }
#pragma unroll
    for (int j = 0; j < 4; ++j) {
        int d = lane + 32 * j;
        float rot = (j < 2) ? -xn[j + 2] : xn[j - 2];  // rotate_half partner
        float y = xn[j] * cosb[(size_t)s * 128 + d] + rot * sinb[(size_t)s * 128 + d];
        Y[(size_t)wid * 128 + d] = (__bf16)y;
    }
}

// ---------------------------------------------------------------------------
// Kernel 5: flash attention (causal, GQA). Block = 4 waves x 16 q rows.
// Double-buffered async K/V tiles; diagonal (masked) block peeled out of the
// steady-state loop so the hot path has no mask VALU and overlapped DMA.
// ---------------------------------------------------------------------------
__global__ __launch_bounds__(128) void flash_kernel(
    const __bf16* __restrict__ Q, const __bf16* __restrict__ Kg,
    const __bf16* __restrict__ Vt,   // (B, NKV, 128, S): [dim][seq] per head
    float* __restrict__ Og,
    int B_, int S_, int NH_, int NKV_) {
    __shared__ __bf16 Ks[2][64][128];  // [key][dim]  B-frag ready for QK^T
    __shared__ __bf16 Vs[2][128][64];  // [dim][key]  B-frag ready for PV
    __shared__ __bf16 Ps[4][16][64];   // per-wave P staging -> A fragments

    const int tid  = threadIdx.x;
    const int lane = tid & 31;
    const int wave = tid >> 5;
    const int half = lane >> 4;
    const int ln   = lane & 15;
    const int lmrow = half * 8;
    const int qt = blockIdx.x, h = blockIdx.y, b = blockIdx.z;
    const int kvh = h / (NH_ / NKV_);
    const float scale = 0.08838834764831845f;  // 1/sqrt(128)

    // Q fragments in registers for the whole kernel (A layout == global layout)
    bf16x16 qf[4];
    {
        int qrow = qt * 64 + wave * 16 + ln;
        const __bf16* qp = Q + (((size_t)b * S_ + qrow) * NH_ + h) * 128;
#pragma unroll
        for (int kc = 0; kc < 4; ++kc) {
            const __bf16* p = qp + kc * 32 + half * 8;
            qf[kc] = make16(*(const bf16x8*)p, *(const bf16x8*)(p + 16));
        }
    }

    // fixed async-copy slots: K tile (8 x b128/thread), V tile (8 x b128/thread)
    const int kr = tid >> 4, dd = (tid & 15) * 8;
    const size_t krow = (size_t)NKV_ * 128;
    const __bf16* kbase = Kg + ((size_t)b * S_ + kr) * krow + (size_t)kvh * 128 + dd;
    unsigned lks[8];
#pragma unroll
    for (int j = 0; j < 8; ++j) lks[j] = lds_addr(&Ks[0][kr + 8 * j][dd]);

    const int vd = tid >> 3, sk = (tid & 7) * 8;
    const __bf16* vbase = Vt + ((size_t)(b * NKV_ + kvh) * 128 + vd) * S_ + sk;
    unsigned lvs[8];
#pragma unroll
    for (int j = 0; j < 8; ++j) lvs[j] = lds_addr(&Vs[0][vd + 16 * j][sk]);

    const unsigned kbuf = (unsigned)(64 * 128 * sizeof(__bf16));   // 16384
    const unsigned vbuf = (unsigned)(128 * 64 * sizeof(__bf16));   // 16384

    auto issue_kv = [&](int kb, int buf) {
        unsigned ok = buf ? kbuf : 0u, ov = buf ? vbuf : 0u;
        const __bf16* kp = kbase + (size_t)kb * 64 * krow;
        const __bf16* vp = vbase + (size_t)kb * 64;
#pragma unroll
        for (int j = 0; j < 8; ++j)
            async_copy_b128(lks[j] + ok, kp + (size_t)(8 * j) * krow);
#pragma unroll
        for (int j = 0; j < 8; ++j)
            async_copy_b128(lvs[j] + ov, vp + (size_t)(16 * j) * S_);
    };

    f32x8 o[8] = {};
    float m_r[8], l_r[8];
#pragma unroll
    for (int r = 0; r < 8; ++r) { m_r[r] = -1e30f; l_r[r] = 0.f; }

    auto process = [&](int kb, int cur, bool diag) {
        // ---- S = Q K^T ----
        f32x8 sacc[4] = {};
#pragma unroll
        for (int kc = 0; kc < 4; ++kc) {
#pragma unroll
            for (int nt = 0; nt < 4; ++nt) {
                bf16x16 bfv = frag_b(&Ks[cur][nt * 16][kc * 32], 128, lane);
                sacc[nt] = wmma_bf16(qf[kc], bfv, sacc[nt]);
            }
        }
        // ---- online softmax (row stats per VGPR slot, per lane-half) ----
        float tmax[8];
#pragma unroll
        for (int r = 0; r < 8; ++r) tmax[r] = -1e30f;
#pragma unroll
        for (int nt = 0; nt < 4; ++nt) {
#pragma unroll
            for (int r = 0; r < 8; ++r) {
                float sv = sacc[nt][r] * scale;
                if (diag) {
                    int kg = kb * 64 + nt * 16 + ln;
                    int qg = qt * 64 + wave * 16 + lmrow + r;
                    if (kg > qg) sv = -1e30f;
                }
                sacc[nt][r] = sv;
                tmax[r] = fmaxf(tmax[r], sv);
            }
        }
#pragma unroll
        for (int r = 0; r < 8; ++r) {
#pragma unroll
            for (int off = 8; off > 0; off >>= 1)   // stays within 16-lane half
                tmax[r] = fmaxf(tmax[r], __shfl_xor(tmax[r], off, 32));
        }
        float alpha[8], rsum[8];
#pragma unroll
        for (int r = 0; r < 8; ++r) {
            float mnew = fmaxf(m_r[r], tmax[r]);
            alpha[r] = __expf(m_r[r] - mnew);
            m_r[r] = mnew;
            rsum[r] = 0.f;
        }
#pragma unroll
        for (int nt = 0; nt < 4; ++nt) {
#pragma unroll
            for (int r = 0; r < 8; ++r) {
                float pv = __expf(sacc[nt][r] - m_r[r]);
                rsum[r] += pv;
                Ps[wave][lmrow + r][nt * 16 + ln] = (__bf16)pv;
            }
        }
#pragma unroll
        for (int r = 0; r < 8; ++r) {
#pragma unroll
            for (int off = 8; off > 0; off >>= 1)
                rsum[r] += __shfl_xor(rsum[r], off, 32);
            l_r[r] = l_r[r] * alpha[r] + rsum[r];
        }
#pragma unroll
        for (int ot = 0; ot < 8; ++ot)
#pragma unroll
            for (int r = 0; r < 8; ++r) o[ot][r] *= alpha[r];

        // ---- O += P V ----
#pragma unroll
        for (int kc2 = 0; kc2 < 2; ++kc2) {
            bf16x16 af = frag_a(&Ps[wave][0][kc2 * 32], 64, lane);
#pragma unroll
            for (int ot = 0; ot < 8; ++ot) {
                bf16x16 bfv = frag_b(&Vs[cur][ot * 16][kc2 * 32], 64, lane);
                o[ot] = wmma_bf16(af, bfv, o[ot]);
            }
        }
    };

    issue_kv(0, 0);
    int cur = 0;
    for (int kb = 0; kb < qt; ++kb) {        // steady state: unmasked blocks
        issue_kv(kb + 1, cur ^ 1);           // overlap next tile's DMA
        WAIT_ASYNC(16);                      // oldest 16 (current tile) done
        __syncthreads();
        process(kb, cur, false);
        __syncthreads();
        cur ^= 1;
    }
    WAIT_ASYNC(0);                           // peeled diagonal (masked) block
    __syncthreads();
    process(qt, cur, true);

    // epilogue: normalize and write ctx f32 (B,S,NH,128)
#pragma unroll
    for (int ot = 0; ot < 8; ++ot) {
#pragma unroll
        for (int r = 0; r < 8; ++r) {
            int srow = qt * 64 + wave * 16 + lmrow + r;
            float val = o[ot][r] / l_r[r];
            Og[(((size_t)b * S_ + srow) * NH_ + h) * 128 + ot * 16 + ln] = val;
        }
    }
}

// ---------------------------------------------------------------------------
// Host-side orchestration
// ---------------------------------------------------------------------------
extern "C" void kernel_launch(void* const* d_in, const int* in_sizes, int n_in,
                              void* d_out, int out_size, void* d_ws, size_t ws_size,
                              hipStream_t stream) {
    (void)in_sizes; (void)n_in; (void)out_size; (void)ws_size;
    constexpr int Bc = 2, S = 2048, H = 2048, NH = 16, NKV = 8;
    constexpr int M = Bc * S;              // 4096 tokens
    constexpr int Nq = NH * 128;           // 2048
    constexpr int Nkv = NKV * 128;         // 1024

    const float* hs  = (const float*)d_in[0];
    const float* wq  = (const float*)d_in[1];
    const float* wk  = (const float*)d_in[2];
    const float* wv  = (const float*)d_in[3];
    const float* wo  = (const float*)d_in[4];
    const float* qnw = (const float*)d_in[5];
    const float* knw = (const float*)d_in[6];
    const float* cs  = (const float*)d_in[7];
    const float* sn  = (const float*)d_in[8];
    float* out = (float*)d_out;

    char* ws = (char*)d_ws;
    size_t off = 0;
    auto take = [&](size_t bytes) -> void* {
        void* p = ws + off;
        off += (bytes + 255) & ~(size_t)255;
        return p;
    };
    __bf16* hsb  = (__bf16*)take((size_t)M * H * 2);
    __bf16* wqT  = (__bf16*)take((size_t)H * Nq * 2);   // (Nq x H)
    __bf16* wkT  = (__bf16*)take((size_t)H * Nkv * 2);  // (Nkv x H)
    __bf16* wvT  = (__bf16*)take((size_t)H * Nkv * 2);  // (Nkv x H)
    __bf16* woT  = (__bf16*)take((size_t)Nq * H * 2);   // (H x Nq)
    float*  qf   = (float*)take((size_t)M * Nq * 4);
    float*  kf   = (float*)take((size_t)M * Nkv * 4);
    float*  vf   = (float*)take((size_t)M * Nkv * 4);
    __bf16* qb   = (__bf16*)take((size_t)M * Nq * 2);
    __bf16* kb2  = (__bf16*)take((size_t)M * Nkv * 2);
    __bf16* vtb  = (__bf16*)take((size_t)M * Nkv * 2);  // (B, NKV, 128, S)
    float*  ctxf = (float*)take((size_t)M * Nq * 4);
    __bf16* ctxb = (__bf16*)take((size_t)M * Nq * 2);

    // 1) convert activations; transpose+convert weights (B operands want [n][k])
    cvt_f32_bf16<<<2048, 256, 0, stream>>>(hs, hsb, (long)M * H);
    transpose_cvt<<<dim3(Nq / 64,  H / 64, 1), 256, 0, stream>>>(wq, wqT, Nq,  H, 0, 0, 1, 0);
    transpose_cvt<<<dim3(Nkv / 64, H / 64, 1), 256, 0, stream>>>(wk, wkT, Nkv, H, 0, 0, 1, 0);
    transpose_cvt<<<dim3(Nkv / 64, H / 64, 1), 256, 0, stream>>>(wv, wvT, Nkv, H, 0, 0, 1, 0);
    transpose_cvt<<<dim3(H / 64, Nq / 64, 1), 256, 0, stream>>>(wo, woT, H, Nq, 0, 0, 1, 0);

    // 2) QKV projections (WMMA GEMMs, async double-buffered tiles)
    gemm_bf16<<<dim3(Nq / BN,  M / BM), 256, 0, stream>>>(hsb, wqT, qf, M, Nq,  H);
    gemm_bf16<<<dim3(Nkv / BN, M / BM), 256, 0, stream>>>(hsb, wkT, kf, M, Nkv, H);
    gemm_bf16<<<dim3(Nkv / BN, M / BM), 256, 0, stream>>>(hsb, wvT, vf, M, Nkv, H);

    // 3) RMSNorm + RoPE on q/k; V transposed per kv-head to (dim x seq) bf16
    norm_rope<<<(M * NH) / 8,  256, 0, stream>>>(qf, qnw, cs, sn, qb, S, NH);
    norm_rope<<<(M * NKV) / 8, 256, 0, stream>>>(kf, knw, cs, sn, kb2, S, NKV);
    transpose_cvt<<<dim3(128 / 64, S / 64, Bc * NKV), 256, 0, stream>>>(
        vf, vtb, (long)NKV * 128, (long)S,
        (long)S * NKV * 128, (long)128, NKV, (long)128 * S);

    // 4) flash attention (causal, GQA), double-buffered async K/V tiles
    flash_kernel<<<dim3(S / 64, NH, Bc), 128, 0, stream>>>(qb, kb2, vtb, ctxf,
                                                           Bc, S, NH, NKV);

    // 5) output projection -> d_out (f32)
    cvt_f32_bf16<<<2048, 256, 0, stream>>>(ctxf, ctxb, (long)M * Nq);
    gemm_bf16<<<dim3(H / BN, M / BM), 256, 0, stream>>>(ctxb, woT, out, M, H, Nq);
}
